// AgentTransformer_20203526160414
// MI455X (gfx1250) — matrix operand order repaired and checked by reading, not compiled, not run on hardware
//
#include <hip/hip_runtime.h>
#include <hip/hip_bf16.h>
#include <math.h>

// ---------------------------------------------------------------------------
// Problem constants (from reference)
// ---------------------------------------------------------------------------
#define B_ 4
#define N_ 3600
#define M_ 3600
#define C_ 384
#define K_ 14
#define DK_ 512
#define ROWS_ (B_ * N_)       // 14400 (N==M so shared)
#define NEGV (-10000.0f)

typedef __attribute__((ext_vector_type(16))) _Float16 v16h;
typedef __attribute__((ext_vector_type(8)))  float    v8f;
typedef __attribute__((ext_vector_type(4)))  float    f32x4;
typedef __attribute__((ext_vector_type(4)))  unsigned int u32x4;
typedef __attribute__((ext_vector_type(8)))  int      i32x8;
typedef __attribute__((ext_vector_type(4)))  int      i32x4;

union HV { f32x4 q[2]; v16h h; };

static __device__ __forceinline__ v8f wmma_f16(v16h a, v16h b, v8f c) {
  // D = A(16x32 f16) x B(32x16 f16) + C(16x16 f32)
  return __builtin_amdgcn_wmma_f32_16x16x32_f16(false, a, false, b, (short)0, c,
                                                false, false);
}

static __device__ __forceinline__ v8f zero8() {
  v8f z = {0.f, 0.f, 0.f, 0.f, 0.f, 0.f, 0.f, 0.f};
  return z;
}

// ---------------------------------------------------------------------------
// Tensor Data Mover: async DMA of a [rows<=32, C_] f16 tile (row-major,
// row stride C_) from global memory into LDS. 2-D descriptor; rows beyond
// rows_avail are zero-filled by the TDM OOB rule. Tracked by TENSORcnt.
// ---------------------------------------------------------------------------
static __device__ __forceinline__ void tdm_load_tile(const _Float16* gsrc,
                                                     _Float16* lds_dst,
                                                     int rows_avail) {
  unsigned lds_off = (unsigned)(size_t)lds_dst;      // low 32 bits = LDS offset
  unsigned long long ga = (unsigned long long)(size_t)gsrc;
  u32x4 g0 = {0u, 0u, 0u, 0u};
  g0[0] = 1u;                                        // count=1, user descriptor
  g0[1] = lds_off;                                   // lds_addr (bytes)
  g0[2] = (unsigned)(ga & 0xFFFFFFFFull);            // global_addr[31:0]
  g0[3] = (unsigned)((ga >> 32) & 0x1FFFFFFull) | (2u << 30); // addr[56:32]|type=2
  i32x8 g1 = {0, 0, 0, 0, 0, 0, 0, 0};
  unsigned td1 = (unsigned)rows_avail;               // tensor_dim1 (rows left)
  g1[0] = (int)(1u << 16);                           // data_size=1 -> 2 bytes
  g1[1] = (int)(((unsigned)C_ & 0xFFFFu) << 16);     // tensor_dim0[15:0]
  g1[2] = (int)((((unsigned)C_ >> 16) & 0xFFFFu) | ((td1 & 0xFFFFu) << 16));
  g1[3] = (int)(((td1 >> 16) & 0xFFFFu) | (((unsigned)C_ & 0xFFFFu) << 16)); // tile_dim0=C_
  g1[4] = 32;                                        // tile_dim1 = 32 rows
  g1[5] = C_;                                        // tensor_dim0_stride
  i32x4 gz = {0, 0, 0, 0};
#if defined(__clang_major__) && (__clang_major__ >= 23)
  i32x8 gz8 = {0, 0, 0, 0, 0, 0, 0, 0};
  __builtin_amdgcn_tensor_load_to_lds(g0, g1, gz, gz, gz8, 0);
#else
  __builtin_amdgcn_tensor_load_to_lds(g0, g1, gz, gz, 0);
#endif
}

// ---------------------------------------------------------------------------
// Elementwise f32 -> f16 convert
// ---------------------------------------------------------------------------
__global__ void k_f32_to_f16(const float* __restrict__ s,
                             _Float16* __restrict__ d, long n) {
  long i = (long)blockIdx.x * blockDim.x + threadIdx.x;
  if (i < n) d[i] = (_Float16)s[i];
}

// W [K x Ncol] row-major f32  ->  WT [Ncol x K] row-major f16 (transposed)
__global__ void k_transpose_f16(const float* __restrict__ W,
                                _Float16* __restrict__ WT, int K, int Ncol) {
  int i = blockIdx.x * blockDim.x + threadIdx.x;
  if (i < K * Ncol) {
    int k = i / Ncol, c = i % Ncol;
    WT[(size_t)c * K + k] = (_Float16)W[(size_t)k * Ncol + c];
  }
}

// ---------------------------------------------------------------------------
// Generic row-major WMMA GEMM: Y[rows,cols] = Xh[rows,Kd] * (WT[cols,Kd])^T
// One wave -> one 16x16 output tile; 8 waves per block (128 output cols).
// K dimension split into two interleaved chains (independent accumulators)
// so consecutive v_wmma issues have no RAW dependency.
// ---------------------------------------------------------------------------
__global__ void __launch_bounds__(256)
k_gemm_wmma(const _Float16* __restrict__ Xh, const _Float16* __restrict__ WT,
            float* __restrict__ Yf, _Float16* __restrict__ Yh, int rows, int Kd,
            int cols) {
  int lane = threadIdx.x & 31, wave = threadIdx.x >> 5;
  int row0 = blockIdx.x * 16;
  int col0 = (blockIdx.y * 8 + wave) * 16;
  if (row0 >= rows || col0 >= cols) return;
  int lm = lane & 15, hi = lane >> 4;
  int khalf = hi * 8;   // A: lanes 0-15 K 0..7/16..23, lanes 16-31 K 8..15/24..31
  int khb = hi * 8;     // B: lanes 0-15 K 0..7/16..23, lanes 16-31 K 8..15/24..31
  const _Float16* pa = Xh + (size_t)(row0 + lm) * Kd;
  const _Float16* pb = WT + (size_t)(col0 + lm) * Kd + khb;
  v8f acc0 = zero8(), acc1 = zero8();
  for (int kt = 0; kt < Kd; kt += 64) {     // Kd % 64 == 0 (384, 512)
    HV A0, B0, A1, B1;
    A0.q[0] = *(const f32x4*)(pa + kt + khalf);
    A0.q[1] = *(const f32x4*)(pa + kt + 16 + khalf);
    B0.q[0] = *(const f32x4*)(pb + kt);
    B0.q[1] = *(const f32x4*)(pb + kt + 16);
    A1.q[0] = *(const f32x4*)(pa + kt + 32 + khalf);
    A1.q[1] = *(const f32x4*)(pa + kt + 48 + khalf);
    B1.q[0] = *(const f32x4*)(pb + kt + 32);
    B1.q[1] = *(const f32x4*)(pb + kt + 48);
    if (kt + 128 < Kd) {
      __builtin_prefetch(pa + kt + 128, 0, 1);
      __builtin_prefetch(pb + kt + 128, 0, 1);
    }
    acc0 = wmma_f16(A0.h, B0.h, acc0);
    acc1 = wmma_f16(A1.h, B1.h, acc1);
  }
#pragma unroll
  for (int j = 0; j < 8; j++) {
    int r = row0 + j + 8 * hi;
    int c = col0 + lm;
    float v = acc0[j] + acc1[j];
    if (Yf) Yf[(size_t)r * cols + c] = v;
    if (Yh) Yh[(size_t)r * cols + c] = (_Float16)v;
  }
}

// ---------------------------------------------------------------------------
// seed = swapaxes(seed_vector,1,2) @ W_s      seed_vector [B,C,K] -> [B,K,C]
// ---------------------------------------------------------------------------
__global__ void k_seed(const float* __restrict__ sv, const float* __restrict__ Ws,
                       float* __restrict__ seed) {
  int i = blockIdx.x * blockDim.x + threadIdx.x;  // B*K*C
  if (i >= B_ * K_ * C_) return;
  int c = i % C_;
  int t = i / C_;
  int kk = t % K_, b = t / K_;
  const float* col = sv + (size_t)b * C_ * K_ + kk;  // stride K_
  float s = 0.f;
  for (int c2 = 0; c2 < C_; c2++) s += col[(size_t)c2 * K_] * Ws[(size_t)c2 * C_ + c];
  seed[i] = s;
}

// qhs[B,K,DK] = seed[B,K,C] @ W_qs[C,DK]
__global__ void k_seedproj(const float* __restrict__ seed,
                           const float* __restrict__ Wqs,
                           float* __restrict__ qhs) {
  int i = blockIdx.x * blockDim.x + threadIdx.x;  // B*K*DK
  if (i >= B_ * K_ * DK_) return;
  int d = i % DK_;
  int t = i / DK_;
  float s = 0.f;
  const float* row = seed + (size_t)t * C_;
  for (int c = 0; c < C_; c++) s += row[c] * Wqs[(size_t)c * DK_ + d];
  qhs[i] = s;
}

// ---------------------------------------------------------------------------
// MHA row: one block per (b, k) of the 14 seed queries. logits -> softmax -> PV
// ---------------------------------------------------------------------------
__global__ void __launch_bounds__(256)
k_mha(const float* __restrict__ qhs, const float* __restrict__ kh,
      const float* __restrict__ vh, const int* __restrict__ smask,
      float* __restrict__ out) {
  __shared__ float sAttn[M_];
  __shared__ float red[256];
  int b = blockIdx.x / K_, kk = blockIdx.x % K_;
  const float* qrow = qhs + ((size_t)b * K_ + kk) * DK_;
  float scale = rsqrtf((float)DK_);
  for (int mm = threadIdx.x; mm < M_; mm += blockDim.x) {
    const float* krow = kh + ((size_t)b * M_ + mm) * DK_;
    float d = 0.f;
    for (int c = 0; c < DK_; c++) d += qrow[c] * krow[c];
    sAttn[mm] = (smask[(size_t)b * M_ + mm] > 0) ? d * scale : NEGV;
  }
  __syncthreads();
  float mx = -3.0e38f;
  for (int mm = threadIdx.x; mm < M_; mm += blockDim.x) mx = fmaxf(mx, sAttn[mm]);
  red[threadIdx.x] = mx;
  __syncthreads();
  for (int s = 128; s > 0; s >>= 1) {
    if (threadIdx.x < s) red[threadIdx.x] = fmaxf(red[threadIdx.x], red[threadIdx.x + s]);
    __syncthreads();
  }
  mx = red[0];
  __syncthreads();
  float sm = 0.f;
  for (int mm = threadIdx.x; mm < M_; mm += blockDim.x) {
    float e = __expf(sAttn[mm] - mx);
    sAttn[mm] = e;
    sm += e;
  }
  red[threadIdx.x] = sm;
  __syncthreads();
  for (int s = 128; s > 0; s >>= 1) {
    if (threadIdx.x < s) red[threadIdx.x] += red[threadIdx.x + s];
    __syncthreads();
  }
  float inv = 1.f / red[0];
  __syncthreads();
  for (int mm = threadIdx.x; mm < M_; mm += blockDim.x) sAttn[mm] *= inv;
  __syncthreads();
  for (int d = threadIdx.x; d < DK_; d += blockDim.x) {
    float s = 0.f;
    for (int mm = 0; mm < M_; mm++)
      s += sAttn[mm] * vh[((size_t)b * M_ + mm) * DK_ + d];
    out[((size_t)b * K_ + kk) * DK_ + d] = s;
  }
}

// out@W_fc + residual -> LayerNorm -> protos. One block (384 thr) per (b,k).
__global__ void __launch_bounds__(384)
k_proto(const float* __restrict__ mo, const float* __restrict__ Wfc,
        const float* __restrict__ seed, const float* __restrict__ lng,
        const float* __restrict__ lnb, float* __restrict__ protos) {
  __shared__ float red[C_];
  __shared__ float st[2];
  int b = blockIdx.x / K_, kk = blockIdx.x % K_;
  int c = threadIdx.x;
  const float* orow = mo + ((size_t)b * K_ + kk) * DK_;
  float s = 0.f;
  for (int d = 0; d < DK_; d++) s += orow[d] * Wfc[(size_t)d * C_ + c];
  float x = s + seed[((size_t)b * K_ + kk) * C_ + c];
  red[c] = x;
  __syncthreads();
  if (c == 0) {
    float mu = 0.f;
    for (int i = 0; i < C_; i++) mu += red[i];
    st[0] = mu / C_;
  }
  __syncthreads();
  float mu = st[0];
  red[c] = (x - mu) * (x - mu);
  __syncthreads();
  if (c == 0) {
    float var = 0.f;
    for (int i = 0; i < C_; i++) var += red[i];
    st[1] = var / C_;
  }
  __syncthreads();
  float inv = rsqrtf(st[1] + 1e-5f);
  protos[((size_t)b * K_ + kk) * C_ + c] = lng[c] * (x - mu) * inv + lnb[c];
}

// pk/pq projection + per-row l2 normalize. One block per (b,k).
__global__ void __launch_bounds__(384)
k_pkpq(const float* __restrict__ protos, const float* __restrict__ Wsk,
       const float* __restrict__ Wsq, float* __restrict__ pkn,
       float* __restrict__ pqn) {
  __shared__ float rk[C_], rq[C_];
  __shared__ float nrm[2];
  int b = blockIdx.x / K_, kk = blockIdx.x % K_;
  int c = threadIdx.x;
  const float* pr = protos + ((size_t)b * K_ + kk) * C_;
  float sk = 0.f, sq = 0.f;
  for (int c2 = 0; c2 < C_; c2++) {
    float p = pr[c2];
    sk += p * Wsk[(size_t)c2 * C_ + c];
    sq += p * Wsq[(size_t)c2 * C_ + c];
  }
  rk[c] = sk * sk;
  rq[c] = sq * sq;
  __syncthreads();
  if (c == 0) {
    float a = 0.f, bb = 0.f;
    for (int i = 0; i < C_; i++) { a += rk[i]; bb += rq[i]; }
    nrm[0] = 1.f / fmaxf(sqrtf(a), 1e-12f);
    nrm[1] = 1.f / fmaxf(sqrtf(bb), 1e-12f);
  }
  __syncthreads();
  pkn[((size_t)b * K_ + kk) * C_ + c] = sk * nrm[0];
  pqn[((size_t)b * K_ + kk) * C_ + c] = sq * nrm[1];
}

// ---------------------------------------------------------------------------
// att_q2p = cos_sim(qf, pk) -> padded f16 [row,32] + argmax id. 1 wave / row.
// ---------------------------------------------------------------------------
__global__ void __launch_bounds__(256)
k_q2p(const float* __restrict__ qf, const float* __restrict__ pkn,
      _Float16* __restrict__ q2ph, float* __restrict__ idq) {
  int lane = threadIdx.x & 31;
  long row = (long)blockIdx.x * 8 + (threadIdx.x >> 5);
  if (row >= ROWS_) return;
  int b = (int)(row / N_);
  const float* qr = qf + row * C_;
  float ss = 0.f;
  for (int c = lane; c < C_; c += 32) { float v = qr[c]; ss += v * v; }
#pragma unroll
  for (int o = 16; o > 0; o >>= 1) ss += __shfl_xor(ss, o, 32);
  float qinv = 1.f / fmaxf(sqrtf(ss), 1e-12f);
  float best = -3.0e38f;
  int bi = 0;
  for (int kk = 0; kk < K_; kk++) {
    const float* pr = pkn + ((size_t)b * K_ + kk) * C_;
    float d = 0.f;
    for (int c = lane; c < C_; c += 32) d += qr[c] * pr[c];
#pragma unroll
    for (int o = 16; o > 0; o >>= 1) d += __shfl_xor(d, o, 32);
    float att = d * qinv;
    if (lane == kk) q2ph[row * 32 + kk] = (_Float16)att;
    if (att > best) { best = att; bi = kk; }
  }
  if (lane >= K_) q2ph[row * 32 + lane] = (_Float16)0.f;  // zero pad K..31
  if (lane == 0) idq[row] = (float)bi;
}

// ---------------------------------------------------------------------------
// OT enhance: cost -> Km -> 100 sinkhorn iterations -> att_s2p (padded f16) + id
// One 1024-thread block per batch. u lives in LDS, Km in global scratch.
// ---------------------------------------------------------------------------
__global__ void __launch_bounds__(1024)
k_ot(const float* __restrict__ kf, const float* __restrict__ pqn,
     const int* __restrict__ smask, float* __restrict__ Kmg,
     _Float16* __restrict__ s2ph, float* __restrict__ ids) {
  __shared__ float u[M_];
  __shared__ float cr[32][16];
  __shared__ float vsh[16];
  __shared__ float sc0[1];
  int b = blockIdx.x;
  int tid = threadIdx.x;
  int lane = tid & 31, wv = tid >> 5;
  const float reg = 0.05f;

  // count background pixels
  float cbg = 0.f;
  for (int mm = tid; mm < M_; mm += 1024) cbg += (smask[(size_t)b * M_ + mm] == 0) ? 1.f : 0.f;
#pragma unroll
  for (int o = 16; o > 0; o >>= 1) cbg += __shfl_xor(cbg, o, 32);
  if (lane == 0) cr[wv][0] = cbg;
  __syncthreads();
  if (tid == 0) {
    float s = 0.f;
    for (int w = 0; w < 32; w++) s += cr[w][0];
    sc0[0] = s;
  }
  __syncthreads();
  float num_bg = sc0[0];
  float num_fg = (float)M_ - num_bg;

  // Km = exp(-cost/reg), cost = [1 - cos_sim, 2*maskf]
  float* Km = Kmg + (size_t)b * M_ * 15;
  for (int mm = tid; mm < M_; mm += 1024) {
    const float* kr = kf + ((size_t)b * M_ + mm) * C_;
    float ss = 0.f;
    for (int c = 0; c < C_; c++) { float x = kr[c]; ss += x * x; }
    float inv = 1.f / fmaxf(sqrtf(ss), 1e-12f);
    float maskf = (smask[(size_t)b * M_ + mm] > 0) ? 1.f : 0.f;
    for (int j = 0; j < K_; j++) {
      const float* pr = pqn + ((size_t)b * K_ + j) * C_;
      float d = 0.f;
      for (int c = 0; c < C_; c++) d += kr[c] * pr[c];
      float att = d * inv;
      Km[(size_t)mm * 15 + j] = __expf(-(1.f - att) / reg);
    }
    Km[(size_t)mm * 15 + 14] = __expf(-(2.f * maskf) / reg);
    u[mm] = 1.f / (float)M_;
  }
  __syncthreads();

  for (int it = 0; it < 100; it++) {
    float part[15];
#pragma unroll
    for (int j = 0; j < 15; j++) part[j] = 0.f;
    for (int mm = tid; mm < M_; mm += 1024) {
      float uu = u[mm];
#pragma unroll
      for (int j = 0; j < 15; j++) part[j] += Km[(size_t)mm * 15 + j] * uu;
    }
#pragma unroll
    for (int j = 0; j < 15; j++) {
#pragma unroll
      for (int o = 16; o > 0; o >>= 1) part[j] += __shfl_xor(part[j], o, 32);
    }
    if (lane == 0) {
#pragma unroll
      for (int j = 0; j < 15; j++) cr[wv][j] = part[j];
    }
    __syncthreads();
    if (tid < 15) {
      float s = 0.f;
      for (int w = 0; w < 32; w++) s += cr[w][tid];
      float rb = (tid < K_) ? (num_fg / (float)K_ / (float)M_) : (num_bg / (float)M_);
      vsh[tid] = rb / (s + 1e-16f);
    }
    __syncthreads();
    for (int mm = tid; mm < M_; mm += 1024) {
      float s = 0.f;
#pragma unroll
      for (int j = 0; j < 15; j++) s += Km[(size_t)mm * 15 + j] * vsh[j];
      u[mm] = (1.f / (float)M_) / (s + 1e-16f);
    }
    __syncthreads();
  }

  // T = u * Km * v * num; relu; drop trash column; argmax
  for (int mm = tid; mm < M_; mm += 1024) {
    float uu = u[mm];
    float best = -3.0e38f;
    int bi = 0;
    for (int j = 0; j < K_; j++) {
      float T = fmaxf(uu * Km[(size_t)mm * 15 + j] * vsh[j] * (float)M_, 0.f);
      s2ph[((size_t)b * M_ + mm) * 32 + j] = (_Float16)T;
      if (T > best) { best = T; bi = j; }
    }
    for (int j = K_; j < 32; j++) s2ph[((size_t)b * M_ + mm) * 32 + j] = (_Float16)0.f;
    ids[(size_t)b * M_ + mm] = (float)bi;
  }
}

// ---------------------------------------------------------------------------
// Fused flash attention: softmax(att_q2p @ att_s2p^T + bias + align*NEG) @ vf
// Block = 256 threads (8 waves) owns a 16-row N tile and all 384 columns.
// V tiles are double-buffered in LDS via the Tensor Data Mover: wave0 issues
// the DMA for tile t+1, waits TENSORcnt<=1 (tile t landed), computes scores +
// online softmax into sP; after the barrier all waves run the P@V WMMAs.
// ---------------------------------------------------------------------------
__global__ void __launch_bounds__(256)
k_flash(const _Float16* __restrict__ q2ph, const _Float16* __restrict__ s2ph,
        const float* __restrict__ idq, const float* __restrict__ ids,
        const int* __restrict__ vmask, const _Float16* __restrict__ vfh,
        _Float16* __restrict__ xh) {
  __shared__ __align__(16) _Float16 sV[2][32 * C_];
  __shared__ __align__(16) _Float16 sP[16 * 32];
  __shared__ float sScale[16];
  __shared__ float sIdq[16];
  __shared__ float sInvL[16];
  int b = blockIdx.y;
  int n0 = blockIdx.x * 16;
  int tid = threadIdx.x, lane = tid & 31, wave = tid >> 5;
  int hi = lane >> 4, lm = lane & 15;

  if (tid < 16) sIdq[tid] = idq[(size_t)b * N_ + n0 + tid];

  HV aQ;
  if (wave == 0) {
    const _Float16* pa = q2ph + ((size_t)b * N_ + n0 + lm) * 32;
    aQ.q[0] = *(const f32x4*)(pa + hi * 16);
    aQ.q[1] = *(const f32x4*)(pa + 8 + hi * 16);
  }

  v8f acc[3];
#pragma unroll
  for (int t = 0; t < 3; t++) acc[t] = zero8();
  float mrow[8], lrow[8];
#pragma unroll
  for (int j = 0; j < 8; j++) { mrow[j] = -3.0e38f; lrow[j] = 0.f; }

  const int NT = (M_ + 31) / 32;  // 113 tiles (last is ragged, TDM zero-fills)
  if (wave == 0)
    tdm_load_tile(vfh + (size_t)b * M_ * C_, &sV[0][0], M_);

  for (int t = 0; t < NT; ++t) {
    int m0 = t * 32;
    __syncthreads();  // prev consumers done; buf[(t+1)&1] is free
    if (wave == 0) {
      // prefetch next V tile via TDM, then ensure tile t has landed
      if (t + 1 < NT) {
        tdm_load_tile(vfh + ((size_t)b * M_ + (size_t)(t + 1) * 32) * C_,
                      &sV[(t + 1) & 1][0], M_ - (t + 1) * 32);
        __builtin_amdgcn_s_wait_tensorcnt(1);
      } else {
        __builtin_amdgcn_s_wait_tensorcnt(0);
      }

      v8f sc0v = zero8(), sc1v = zero8();
      int mc0 = m0 + lm, mc1 = m0 + 16 + lm;
      int mr0 = (mc0 < M_) ? mc0 : (M_ - 1);
      int mr1 = (mc1 < M_) ? mc1 : (M_ - 1);
      {
        HV B0, B1;
        const _Float16* pb0 = s2ph + ((size_t)b * M_ + mr0) * 32 + hi * 16;
        const _Float16* pb1 = s2ph + ((size_t)b * M_ + mr1) * 32 + hi * 16;
        B0.q[0] = *(const f32x4*)(pb0);
        B0.q[1] = *(const f32x4*)(pb0 + 8);
        B1.q[0] = *(const f32x4*)(pb1);
        B1.q[1] = *(const f32x4*)(pb1 + 8);
        sc0v = wmma_f16(aQ.h, B0.h, sc0v);
        sc1v = wmma_f16(aQ.h, B1.h, sc1v);
      }
      bool oob0 = mc0 >= M_, oob1 = mc1 >= M_;
      float v0 = oob0 ? 0.f : (float)vmask[(size_t)b * M_ + mc0];
      float v1 = oob1 ? 0.f : (float)vmask[(size_t)b * M_ + mc1];
      float is0 = oob0 ? 0.f : ids[(size_t)b * M_ + mc0];
      float is1 = oob1 ? 0.f : ids[(size_t)b * M_ + mc1];
#pragma unroll
      for (int j = 0; j < 8; j++) {
        int r = j + 8 * hi;
        float iq = sIdq[r];
        float s0 = sc0v[j], s1 = sc1v[j];
        float t0 = 2.f * sqrtf((is0 + 1.5f) * (iq + 1.5f));
        float t1 = 2.f * sqrtf((is1 + 1.5f) * (iq + 1.5f));
        float na0 = ((t0 - floorf(t0)) == 0.f) ? 0.f : 1.f;
        float na1 = ((t1 - floorf(t1)) == 0.f) ? 0.f : 1.f;
        s0 += (v0 + na0) * NEGV;
        s1 += (v1 + na1) * NEGV;
        if (oob0) s0 = -3.0e38f;
        if (oob1) s1 = -3.0e38f;
        float mx = fmaxf(s0, s1);
#pragma unroll
        for (int o = 1; o < 16; o <<= 1) mx = fmaxf(mx, __shfl_xor(mx, o, 32));
        float nm = fmaxf(mrow[j], mx);
        float scale = __expf(mrow[j] - nm);
        float p0 = __expf(s0 - nm);
        float p1 = __expf(s1 - nm);
        float rs = p0 + p1;
#pragma unroll
        for (int o = 1; o < 16; o <<= 1) rs += __shfl_xor(rs, o, 32);
        lrow[j] = lrow[j] * scale + rs;
        mrow[j] = nm;
        sP[r * 32 + lm] = (_Float16)p0;
        sP[r * 32 + 16 + lm] = (_Float16)p1;
        if (lm == 0) sScale[r] = scale;
      }
    }
    __syncthreads();

    // all waves: rescale accumulators, load P (A-frag) and V cols (B-frag)
    const _Float16* sVb = &sV[t & 1][0];
    HV aP;
    {
      const _Float16* pp = sP + lm * 32 + hi * 16;
      aP.q[0] = *(const f32x4*)(pp);
      aP.q[1] = *(const f32x4*)(pp + 8);
    }
#pragma unroll
    for (int j = 0; j < 8; j++) {
      float sc = sScale[j + 8 * hi];
      acc[0][j] *= sc;
      acc[1][j] *= sc;
      acc[2][j] *= sc;
    }
#pragma unroll
    for (int t2 = 0; t2 < 3; t2++) {
      int c0 = (wave * 3 + t2) * 16 + lm;
      HV Bv;
#pragma unroll
      for (int i = 0; i < 8; i++) {
        Bv.h[i] = sVb[(hi * 16 + i) * C_ + c0];
        Bv.h[8 + i] = sVb[(hi * 16 + 8 + i) * C_ + c0];
      }
      acc[t2] = wmma_f16(aP.h, Bv.h, acc[t2]);
    }
  }

  if (wave == 0) {
#pragma unroll
    for (int j = 0; j < 8; j++)
      if (lm == 0) sInvL[j + 8 * hi] = 1.f / lrow[j];
  }
  __syncthreads();
#pragma unroll
  for (int j = 0; j < 8; j++) {
    int r = j + 8 * hi;
    float il = sInvL[r];
    size_t base = ((size_t)b * N_ + n0 + r) * C_;
#pragma unroll
    for (int t2 = 0; t2 < 3; t2++)
      xh[base + (wave * 3 + t2) * 16 + lm] = (_Float16)(acc[t2][j] * il);
  }
}

// ---------------------------------------------------------------------------
// Host launcher
// ---------------------------------------------------------------------------
extern "C" void kernel_launch(void* const* d_in, const int* in_sizes, int n_in,
                              void* d_out, int out_size, void* d_ws,
                              size_t ws_size, hipStream_t stream) {
  (void)in_sizes; (void)n_in; (void)out_size; (void)ws_size;
  const float* q   = (const float*)d_in[0];
  const float* k   = (const float*)d_in[1];
  const float* v   = (const float*)d_in[2];
  const float* sv  = (const float*)d_in[3];
  const int* vmask = (const int*)d_in[4];
  const int* smask = (const int*)d_in[5];
  const float* W_q = (const float*)d_in[6];
  const float* W_k = (const float*)d_in[7];
  const float* W_v = (const float*)d_in[8];
  const float* W_s = (const float*)d_in[9];
  const float* W_seed_q = (const float*)d_in[10];
  const float* W_seed_k = (const float*)d_in[11];
  const float* W_proj   = (const float*)d_in[12];
  const float* W_qs = (const float*)d_in[13];
  const float* W_ks = (const float*)d_in[14];
  const float* W_vs = (const float*)d_in[15];
  const float* W_fc = (const float*)d_in[16];
  const float* ln_g = (const float*)d_in[17];
  const float* ln_b = (const float*)d_in[18];
  float* out = (float*)d_out;

  // workspace carve-out (256B aligned)
  char* base = (char*)d_ws;
  size_t off = 0;
  auto A = [&](size_t bytes) -> void* {
    void* r = base + off;
    off = (off + bytes + 255) & ~(size_t)255;
    return r;
  };
  const size_t NC = (size_t)ROWS_ * C_;
  const size_t ND = (size_t)ROWS_ * DK_;
  _Float16* q16 = (_Float16*)A(NC * 2);
  _Float16* k16 = (_Float16*)A(NC * 2);
  _Float16* v16 = (_Float16*)A(NC * 2);
  _Float16* WqT = (_Float16*)A((size_t)C_ * C_ * 2);
  _Float16* WkT = (_Float16*)A((size_t)C_ * C_ * 2);
  _Float16* WvT = (_Float16*)A((size_t)C_ * C_ * 2);
  _Float16* WpT = (_Float16*)A((size_t)C_ * C_ * 2);
  _Float16* WksT = (_Float16*)A((size_t)C_ * DK_ * 2);
  _Float16* WvsT = (_Float16*)A((size_t)C_ * DK_ * 2);
  float* qf  = (float*)A(NC * 4);
  float* kf  = (float*)A(NC * 4);
  _Float16* kf16 = (_Float16*)A(NC * 2);
  _Float16* vf16 = (_Float16*)A(NC * 2);
  float* khp = (float*)A(ND * 4);
  float* vhp = (float*)A(ND * 4);
  float* seed   = (float*)A((size_t)B_ * K_ * C_ * 4);
  float* qhs    = (float*)A((size_t)B_ * K_ * DK_ * 4);
  float* mhao   = (float*)A((size_t)B_ * K_ * DK_ * 4);
  float* protos = (float*)A((size_t)B_ * K_ * C_ * 4);
  float* pkn    = (float*)A((size_t)B_ * K_ * C_ * 4);
  float* pqn    = (float*)A((size_t)B_ * K_ * C_ * 4);
  _Float16* q2ph = (_Float16*)A((size_t)ROWS_ * 32 * 2);
  float* idq = (float*)A((size_t)ROWS_ * 4);
  float* Kmg = (float*)A((size_t)B_ * M_ * 15 * 4);
  _Float16* s2ph = (_Float16*)A((size_t)ROWS_ * 32 * 2);
  float* ids = (float*)A((size_t)ROWS_ * 4);
  _Float16* xh16 = (_Float16*)A(NC * 2);

  const long nElem = (long)NC;
  int cvtBlocks = (int)((nElem + 255) / 256);
  k_f32_to_f16<<<cvtBlocks, 256, 0, stream>>>(q, q16, nElem);
  k_f32_to_f16<<<cvtBlocks, 256, 0, stream>>>(k, k16, nElem);
  k_f32_to_f16<<<cvtBlocks, 256, 0, stream>>>(v, v16, nElem);

  int tCC = (C_ * C_ + 255) / 256, tCD = (C_ * DK_ + 255) / 256;
  k_transpose_f16<<<tCC, 256, 0, stream>>>(W_q, WqT, C_, C_);
  k_transpose_f16<<<tCC, 256, 0, stream>>>(W_k, WkT, C_, C_);
  k_transpose_f16<<<tCC, 256, 0, stream>>>(W_v, WvT, C_, C_);
  k_transpose_f16<<<tCC, 256, 0, stream>>>(W_proj, WpT, C_, C_);
  k_transpose_f16<<<tCD, 256, 0, stream>>>(W_ks, WksT, C_, DK_);
  k_transpose_f16<<<tCD, 256, 0, stream>>>(W_vs, WvsT, C_, DK_);

  dim3 gCC(ROWS_ / 16, C_ / 128);
  dim3 gCD(ROWS_ / 16, DK_ / 128);
  k_gemm_wmma<<<gCC, 256, 0, stream>>>(q16, WqT, qf, nullptr, ROWS_, C_, C_);
  k_gemm_wmma<<<gCC, 256, 0, stream>>>(k16, WkT, kf, kf16, ROWS_, C_, C_);
  k_gemm_wmma<<<gCC, 256, 0, stream>>>(v16, WvT, nullptr, vf16, ROWS_, C_, C_);

  k_seed<<<(B_ * K_ * C_ + 255) / 256, 256, 0, stream>>>(sv, W_s, seed);
  k_seedproj<<<(B_ * K_ * DK_ + 255) / 256, 256, 0, stream>>>(seed, W_qs, qhs);
  k_gemm_wmma<<<gCD, 256, 0, stream>>>(kf16, WksT, khp, nullptr, ROWS_, C_, DK_);
  k_gemm_wmma<<<gCD, 256, 0, stream>>>(vf16, WvsT, vhp, nullptr, ROWS_, C_, DK_);

  k_mha<<<B_ * K_, 256, 0, stream>>>(qhs, khp, vhp, smask, mhao);
  k_proto<<<B_ * K_, 384, 0, stream>>>(mhao, W_fc, seed, ln_g, ln_b, protos);
  k_pkpq<<<B_ * K_, 384, 0, stream>>>(protos, W_seed_k, W_seed_q, pkn, pqn);

  k_q2p<<<ROWS_ / 8, 256, 0, stream>>>(qf, pkn, q2ph, idq);
  k_ot<<<B_, 1024, 0, stream>>>(kf, pqn, smask, Kmg, s2ph, ids);

  dim3 gF(N_ / 16, B_);
  k_flash<<<gF, 256, 0, stream>>>(q2ph, s2ph, idq, ids, vmask, vf16, xh16);

  k_gemm_wmma<<<gCC, 256, 0, stream>>>(xh16, WpT, out, nullptr, ROWS_, C_, C_);
}
